// DropConnectLinear_27719718928915
// MI455X (gfx1250) — compile-verified, matrix operand order checked by reading, not run on hardware
//
#include <hip/hip_runtime.h>
#include <stdint.h>

// DropConnect linear for MI455X (gfx1250, wave32).
// y[b,o] = sum_i x[b,i]*w[o,i]*!wmask[b,o,i] + (!bmask[b,o])*bias[o]
//
// Memory-bound on the 268MB per-sample weight mask (~11.5us @ 23.3TB/s).
// One wave owns a 16(b)x16(o) tile. v_wmma_f32_16x16x32_f16 performs the
// K-reduction into a column-selected accumulator; VALU only builds masked
// f16 products. Manual 1-deep software pipeline: iteration i+1's 5 b128
// loads are issued before iteration i's compute, hiding memory latency
// within the wave (w/mask rows are contiguous across the whole 16-o sweep,
// so prefetch pointers advance uniformly with no per-oo pipeline restart).

typedef __attribute__((ext_vector_type(16))) _Float16 v16h;
typedef __attribute__((ext_vector_type(2)))  _Float16 h2;
typedef __attribute__((ext_vector_type(8)))  float    v8f;

#define B_DIM   256
#define IN_DIM  1024
#define OUT_DIM 1024
#define WAVES_PER_BLOCK 8   // 256 threads = 8 wave32s, one 16x16 tile each

// ---- Prepass: f32 -> f16 (removes cvts from hot loop, halves x/w bytes) ----
__global__ __launch_bounds__(256) void cvt_f16_kernel(
    const float* __restrict__ src, _Float16* __restrict__ dst) {
  int i = blockIdx.x * blockDim.x + threadIdx.x;       // 4 floats per thread
  float4 v = ((const float4*)src)[i];
  h2 a = {(_Float16)v.x, (_Float16)v.y};
  h2 b = {(_Float16)v.z, (_Float16)v.w};
  ((h2*)dst)[2 * i + 0] = a;
  ((h2*)dst)[2 * i + 1] = b;
}

__global__ __launch_bounds__(256) void dropconnect_wmma_kernel(
    const _Float16* __restrict__ xh,     // [B, IN]   f16 (prepass)
    const _Float16* __restrict__ wh,     // [OUT, IN] f16 (prepass)
    const float*    __restrict__ bias,   // [OUT]
    const uint8_t*  __restrict__ wmask,  // [B, OUT, IN] bool (1 = drop)
    const uint8_t*  __restrict__ bmask,  // [B, OUT]     bool (1 = drop)
    float*          __restrict__ y)      // [B, OUT]
{
  const int lane = threadIdx.x & 31;
  const int wave = threadIdx.x >> 5;
  const int b0 = blockIdx.x * 16;
  const int o0 = (blockIdx.y * WAVES_PER_BLOCK + wave) * 16;

  const int arow = lane & 15;        // A-matrix row = b index
  const int kb   = (lane >> 4) * 16; // this lane's contiguous i-segment base

  const _Float16* xrow = xh + (size_t)(b0 + arow) * IN_DIM + kb;
  // w rows o0..o0+15 are contiguous: pw advances by 32 halves per iteration
  const _Float16* pw = wh + (size_t)o0 * IN_DIM + kb;
  // mask rows (b, o0..o0+15) are contiguous: pm advances by 32B per iteration
  const uint8_t* pm =
      wmask + ((size_t)(b0 + arow) * OUT_DIM + (size_t)o0) * IN_DIM + kb;

  // ---- pipeline preload (oo=0, chunk 0) ----
  uint4 cxa = *(const uint4*)(xrow + 0);
  uint4 cxb = *(const uint4*)(xrow + 8);
  uint4 cwa = *(const uint4*)(pw + 0);
  uint4 cwb = *(const uint4*)(pw + 8);
  uint4 cmv = *(const uint4*)(pm);

  v8f acc = {};  // 16x16 f32 C/D tile (8 VGPRs), shared across all 16 oo

#pragma unroll 1   // MUST stay rolled: exactly one live Bm (8 VGPRs)
  for (int oo = 0; oo < 16; ++oo) {
    // B (f16 32x16): ones only in column n==oo -> WMMA reduces over K and
    // accumulates into column oo of acc. Built once per 32 inner iters.
    _Float16 bval = (arow == oo) ? (_Float16)1.0f : (_Float16)0.0f;
    v16h Bm;
#pragma unroll
    for (int q = 0; q < 16; ++q) Bm[q] = bval;

#pragma unroll 4
    for (int c = 0; c < 32; ++c) {
      // ---- prefetch next iteration (issued before current compute) ----
      const int xoff = ((c + 1) & 31) * 32;   // x wraps to row start per oo
      // w/mask advance contiguously across the whole 16-o sweep; only the
      // very last prefetch of the kernel is clamped (uniform scalar select)
      const size_t adv = ((oo == 15) && (c == 31)) ? (size_t)0 : (size_t)32;
      uint4 nxa = *(const uint4*)(xrow + xoff);
      uint4 nxb = *(const uint4*)(xrow + xoff + 8);
      uint4 nwa = *(const uint4*)(pw + adv);
      uint4 nwb = *(const uint4*)(pw + adv + 8);
      uint4 nmv = *(const uint4*)(pm + adv);

      // ---- compute current ----
      uint32_t xd[8] = {cxa.x, cxa.y, cxa.z, cxa.w, cxb.x, cxb.y, cxb.z, cxb.w};
      uint32_t wd[8] = {cwa.x, cwa.y, cwa.z, cwa.w, cwb.x, cwb.y, cwb.z, cwb.w};

      // bool(1=drop) -> keep flag -> 0x00/0xFF per byte (carry-free mul)
      uint32_t f[4];
      f[0] = (cmv.x ^ 0x01010101u) * 0xFFu;
      f[1] = (cmv.y ^ 0x01010101u) * 0xFFu;
      f[2] = (cmv.z ^ 0x01010101u) * 0xFFu;
      f[3] = (cmv.w ^ 0x01010101u) * 0xFFu;

      union { v16h v; uint32_t u[8]; } A;
#pragma unroll
      for (int p = 0; p < 8; ++p) {
        uint32_t src = f[p >> 1];
        // expand keep bytes into 16-bit masks for the two f16 halves
        uint32_t sel = (p & 1) ? 0x03030202u : 0x01010000u;
        uint32_t km  = __builtin_amdgcn_perm(src, src, sel);  // v_perm_b32
        h2 prod = __builtin_bit_cast(h2, xd[p]) *
                  __builtin_bit_cast(h2, wd[p]);              // v_pk_mul_f16
        A.u[p] = __builtin_bit_cast(uint32_t, prod) & km;     // zero dropped
      }

      acc = __builtin_amdgcn_wmma_f32_16x16x32_f16(
          /*neg_a=*/false, A.v, /*neg_b=*/false, Bm,
          /*c_mod=*/(short)0, acc, /*reuse_a=*/false, /*reuse_b=*/false);

      // ---- rotate pipeline ----
      cxa = nxa; cxb = nxb; cwa = nwa; cwb = nwb; cmv = nmv;
      pw += 32;
      pm += 32;
    }
  }

  // ---- Epilogue: C/D layout lane->(m,n): n = lane&15, m = r + 8*(lane>>4)
  const int n = lane & 15;
  const int mrow2 = (lane >> 4) * 8;
  const int o = o0 + n;
  const float bv = bias[o];
#pragma unroll
  for (int r = 0; r < 8; ++r) {
    const int b = b0 + mrow2 + r;
    float v = acc[r] + (bmask[(size_t)b * OUT_DIM + o] ? 0.0f : bv);
    y[(size_t)b * OUT_DIM + o] = v;
  }
}

extern "C" void kernel_launch(void* const* d_in, const int* in_sizes, int n_in,
                              void* d_out, int out_size, void* d_ws, size_t ws_size,
                              hipStream_t stream) {
  const float*   x     = (const float*)d_in[0];    // [256,1024]
  const float*   w     = (const float*)d_in[1];    // [1024,1024]
  const float*   bias  = (const float*)d_in[2];    // [1024]
  const uint8_t* wmask = (const uint8_t*)d_in[3];  // [256,1024,1024] bool
  const uint8_t* bmask = (const uint8_t*)d_in[4];  // [256,1024] bool
  float* y = (float*)d_out;

  // Workspace layout (needs 2.5MB): wh [1M halves] | xh [256K halves]
  _Float16* wh = (_Float16*)d_ws;
  _Float16* xhp = wh + (size_t)OUT_DIM * IN_DIM;

  cvt_f16_kernel<<<(OUT_DIM * IN_DIM / 4) / 256, 256, 0, stream>>>(w, wh);
  cvt_f16_kernel<<<(B_DIM * IN_DIM / 4) / 256, 256, 0, stream>>>(x, xhp);

  dim3 grid(B_DIM / 16, OUT_DIM / (16 * WAVES_PER_BLOCK));  // 16 x 8
  dim3 block(32 * WAVES_PER_BLOCK);                          // 256 = 8 waves
  dropconnect_wmma_kernel<<<grid, block, 0, stream>>>(xhp, wh, bias, wmask,
                                                      bmask, y);
}
